// LineClassifier_2748779070289
// MI455X (gfx1250) — matrix operand order repaired and checked by reading, not compiled, not run on hardware
//
#include <hip/hip_runtime.h>
#include <hip/hip_bf16.h>

typedef __attribute__((ext_vector_type(16))) _Float16 v16h;
typedef __attribute__((ext_vector_type(8)))  _Float16 v8h;
typedef __attribute__((ext_vector_type(8)))  float    v8f;

#define HW      16384   // 128*128
#define CIN     256
#define DLOI    128
#define NLINES  16000
#define DFC     1024
#define NPTS0   32
#define NPTS1   8

// ---------------------------------------------------------------------------
// K0: 1x1 conv + ReLU, output channel-last loi[pixel][128] (fp32, L2 resident)
// block = 256 thr: 64 pixels x 4 output-groups(32 ch each); weights tiled in LDS
// ---------------------------------------------------------------------------
__global__ void conv1x1_relu(const float* __restrict__ F, const float* __restrict__ W,
                             const float* __restrict__ B, float* __restrict__ loi) {
    __shared__ float ws[64 * 128];
    int tid = threadIdx.x;
    int pix = tid & 63;
    int og  = tid >> 6;                 // 0..3, 32 outputs each
    int p   = blockIdx.x * 64 + pix;
    float acc[32];
#pragma unroll
    for (int i = 0; i < 32; ++i) acc[i] = 0.0f;
    for (int cb = 0; cb < CIN; cb += 64) {
        __syncthreads();
        for (int i = tid; i < 64 * 128; i += 256) {
            int cl = i >> 7, o = i & 127;
            ws[i] = W[o * CIN + cb + cl];
        }
        __syncthreads();
        for (int cl = 0; cl < 64; ++cl) {
            float f = F[(size_t)(cb + cl) * HW + p];
            const float* wr = ws + cl * 128 + og * 32;
#pragma unroll
            for (int i = 0; i < 32; ++i) acc[i] = fmaf(f, wr[i], acc[i]);
        }
    }
    float* outp = loi + (size_t)p * DLOI + og * 32;
#pragma unroll
    for (int i = 0; i < 32; ++i) outp[i] = fmaxf(acc[i] + B[og * 32 + i], 0.0f);
}

// ---------------------------------------------------------------------------
// K1: transpose-convert w(N=1024 x K=1024 fp32) -> wt[k][n] f16
// ---------------------------------------------------------------------------
__global__ void transpose_to_f16(const float* __restrict__ w, _Float16* __restrict__ wt) {
    int idx = blockIdx.x * 256 + threadIdx.x;
    int n = idx & 1023;
    int k = idx >> 10;
    wt[(size_t)k * 1024 + n] = (_Float16)w[(size_t)n * 1024 + k];
}

// ---------------------------------------------------------------------------
// K2: line pool. block = 1 line, 8 waves; wave j does 4 sample points (max-pool),
// lanes cover 128 ch (4 each). Bilinear reads from channel-last loi (L2-hot).
// Output feat f16 with n = c*8 + j (matches reference reshape), staged in LDS
// for a coalesced 2KB row store.
// ---------------------------------------------------------------------------
__global__ void line_pool_kernel(const float* __restrict__ loi, const float* __restrict__ lines,
                                 _Float16* __restrict__ feat) {
    __shared__ __align__(16) _Float16 srow[DFC];
    int l    = blockIdx.x;
    int tid  = threadIdx.x;
    int lane = tid & 31;
    int wave = tid >> 5;
    float Ux = lines[l * 4 + 0], Uy = lines[l * 4 + 1];
    float Vx = lines[l * 4 + 2], Vy = lines[l * 4 + 3];
    int c0 = lane * 4;
    float m0 = -1e30f, m1 = -1e30f, m2 = -1e30f, m3 = -1e30f;
#pragma unroll
    for (int s = 0; s < 4; ++s) {
        int j   = wave * 4 + s;
        float t = (float)j * (1.0f / 31.0f);
        float px = Ux * t + Vx * (1.0f - t) - 0.5f;
        float py = Uy * t + Vy * (1.0f - t) - 0.5f;
        float fx0 = fminf(fmaxf(floorf(px), 0.0f), 127.0f);
        float fy0 = fminf(fmaxf(floorf(py), 0.0f), 127.0f);
        float fx1 = fminf(fx0 + 1.0f, 127.0f);
        float fy1 = fminf(fy0 + 1.0f, 127.0f);
        int ix0 = (int)fx0, iy0 = (int)fy0, ix1 = (int)fx1, iy1 = (int)fy1;
        float w00 = (fy1 - py) * (fx1 - px);
        float w10 = (py - fy0) * (fx1 - px);
        float w01 = (fy1 - py) * (px - fx0);
        float w11 = (py - fy0) * (px - fx0);
        float4 a = *(const float4*)(loi + ((size_t)(iy0 * 128 + ix0) * DLOI + c0));
        float4 b = *(const float4*)(loi + ((size_t)(iy1 * 128 + ix0) * DLOI + c0));
        float4 c = *(const float4*)(loi + ((size_t)(iy0 * 128 + ix1) * DLOI + c0));
        float4 d = *(const float4*)(loi + ((size_t)(iy1 * 128 + ix1) * DLOI + c0));
        m0 = fmaxf(m0, w00 * a.x + w10 * b.x + w01 * c.x + w11 * d.x);
        m1 = fmaxf(m1, w00 * a.y + w10 * b.y + w01 * c.y + w11 * d.y);
        m2 = fmaxf(m2, w00 * a.z + w10 * b.z + w01 * c.z + w11 * d.z);
        m3 = fmaxf(m3, w00 * a.w + w10 * b.w + w01 * c.w + w11 * d.w);
    }
    srow[(c0 + 0) * 8 + wave] = (_Float16)m0;
    srow[(c0 + 1) * 8 + wave] = (_Float16)m1;
    srow[(c0 + 2) * 8 + wave] = (_Float16)m2;
    srow[(c0 + 3) * 8 + wave] = (_Float16)m3;
    __syncthreads();
    ((uint2*)(feat + (size_t)l * DFC))[tid] = ((const uint2*)srow)[tid];
}

// ---------------------------------------------------------------------------
// K3/K4: GEMM  out = relu(A(16000xK) @ Wt(KxN) + bias), f16 in, f32 acc, f16 out
// per-wave 32Mx32N (4 WMMA acc), block 8 waves = 64Mx128N, K=N=1024.
// ---------------------------------------------------------------------------
__global__ void gemm_relu_f16(const _Float16* __restrict__ A, const _Float16* __restrict__ Wt,
                              const float* __restrict__ bias, _Float16* __restrict__ out16) {
    const int K = 1024, N = 1024;
    int tid   = threadIdx.x;
    int lane  = tid & 31;
    int wave  = tid >> 5;
    int waveM = wave & 1;
    int waveN = wave >> 1;
    int m0 = blockIdx.x * 64 + waveM * 32;
    int n0 = blockIdx.y * 128 + waveN * 32;
    int lm = lane & 15;
    int lh = lane >> 4;   // 0 or 1

    v8f zero = {0.f, 0.f, 0.f, 0.f, 0.f, 0.f, 0.f, 0.f};
    v8f acc[2][2];
    acc[0][0] = zero; acc[0][1] = zero; acc[1][0] = zero; acc[1][1] = zero;

    for (int k0 = 0; k0 < K; k0 += 32) {
        // A tiles: lane<16 holds row m0+lm, K {0..7,16..23}; lane>=16 same row, K {8..15,24..31}
        union { v16h v; v8h h[2]; } a0u, a1u;
        const _Float16* pa0 = A + (size_t)(m0 + lm) * K + k0 + lh * 8;
        a0u.h[0] = *(const v8h*)pa0;
        a0u.h[1] = *(const v8h*)(pa0 + 16);
        const _Float16* pa1 = pa0 + (size_t)16 * K;
        a1u.h[0] = *(const v8h*)pa1;
        a1u.h[1] = *(const v8h*)(pa1 + 16);
        // B tiles: lane holds col n0+lm, K {0..7,16..23} (lane<16) / {8..15,24..31} (lane>=16)
        v16h b0, b1;
#pragma unroll
        for (int s = 0; s < 16; ++s) {
            int kk = (s < 8) ? (lh * 8 + s) : (16 + lh * 8 + (s - 8));
            const _Float16* pb = Wt + (size_t)(k0 + kk) * N + n0 + lm;
            b0[s] = pb[0];
            b1[s] = pb[16];
        }

        acc[0][0] = __builtin_amdgcn_wmma_f32_16x16x32_f16(false, a0u.v, false, b0, (short)0, acc[0][0], false, false);
        acc[0][1] = __builtin_amdgcn_wmma_f32_16x16x32_f16(false, a0u.v, false, b1, (short)0, acc[0][1], false, false);
        acc[1][0] = __builtin_amdgcn_wmma_f32_16x16x32_f16(false, a1u.v, false, b0, (short)0, acc[1][0], false, false);
        acc[1][1] = __builtin_amdgcn_wmma_f32_16x16x32_f16(false, a1u.v, false, b1, (short)0, acc[1][1], false, false);
    }
    // C/D layout: VGPR r -> row r (lanes 0-15) / row r+8 (lanes 16-31), col = lane&15
#pragma unroll
    for (int i = 0; i < 2; ++i) {
#pragma unroll
        for (int j = 0; j < 2; ++j) {
            int col = n0 + j * 16 + lm;
            float bv = bias[col];
#pragma unroll
            for (int r = 0; r < 8; ++r) {
                int row = m0 + i * 16 + lh * 8 + r;
                float v = fmaxf(acc[i][j][r] + bv, 0.0f);
                out16[(size_t)row * N + col] = (_Float16)v;
            }
        }
    }
}

// ---------------------------------------------------------------------------
// K5: final FC (N_LABELS=3): wave per line, cross-lane reduce.
// ---------------------------------------------------------------------------
__global__ void fc3_kernel(const _Float16* __restrict__ x2, const float* __restrict__ w3,
                           const float* __restrict__ b3, float* __restrict__ out) {
    int lane = threadIdx.x & 31;
    int wave = threadIdx.x >> 5;
    int l = blockIdx.x * 8 + wave;
    float a0 = 0.f, a1 = 0.f, a2 = 0.f;
#pragma unroll
    for (int i = 0; i < 32; ++i) {
        int k = lane + 32 * i;
        float v = (float)x2[(size_t)l * DFC + k];
        a0 = fmaf(v, w3[k], a0);
        a1 = fmaf(v, w3[DFC + k], a1);
        a2 = fmaf(v, w3[2 * DFC + k], a2);
    }
    for (int off = 16; off > 0; off >>= 1) {
        a0 += __shfl_xor(a0, off);
        a1 += __shfl_xor(a1, off);
        a2 += __shfl_xor(a2, off);
    }
    if (lane == 0) {
        out[l * 3 + 0] = a0 + b3[0];
        out[l * 3 + 1] = a1 + b3[1];
        out[l * 3 + 2] = a2 + b3[2];
    }
}

// ---------------------------------------------------------------------------
extern "C" void kernel_launch(void* const* d_in, const int* in_sizes, int n_in,
                              void* d_out, int out_size, void* d_ws, size_t ws_size,
                              hipStream_t stream) {
    const float* features = (const float*)d_in[0];
    const float* lines    = (const float*)d_in[1];
    const float* w_fc1    = (const float*)d_in[2];
    const float* b_fc1    = (const float*)d_in[3];
    const float* w1       = (const float*)d_in[4];
    const float* b1       = (const float*)d_in[5];
    const float* w2       = (const float*)d_in[6];
    const float* b2       = (const float*)d_in[7];
    const float* w3       = (const float*)d_in[8];
    const float* b3       = (const float*)d_in[9];
    float* out = (float*)d_out;

    char* base = (char*)d_ws;
    size_t off = 0;
    float*     loi  = (float*)(base + off);     off += (size_t)HW * DLOI * 4;       // 8.39 MB
    _Float16*  wt1  = (_Float16*)(base + off);  off += (size_t)DFC * DFC * 2;       // 2 MB
    _Float16*  wt2  = (_Float16*)(base + off);  off += (size_t)DFC * DFC * 2;       // 2 MB
    _Float16*  feat = (_Float16*)(base + off);  off += (size_t)NLINES * DFC * 2;    // 32.8 MB
    _Float16*  x1   = (_Float16*)(base + off);  off += (size_t)NLINES * DFC * 2;    // 32.8 MB
    _Float16*  x2   = (_Float16*)(base + off);  off += (size_t)NLINES * DFC * 2;    // 32.8 MB

    conv1x1_relu<<<HW / 64, 256, 0, stream>>>(features, w_fc1, b_fc1, loi);
    transpose_to_f16<<<(DFC * DFC) / 256, 256, 0, stream>>>(w1, wt1);
    transpose_to_f16<<<(DFC * DFC) / 256, 256, 0, stream>>>(w2, wt2);
    line_pool_kernel<<<NLINES, 256, 0, stream>>>(loi, lines, feat);
    gemm_relu_f16<<<dim3(NLINES / 64, DFC / 128), 256, 0, stream>>>(feat, wt1, b1, x1);
    gemm_relu_f16<<<dim3(NLINES / 64, DFC / 128), 256, 0, stream>>>(x1, wt2, b2, x2);
    fc3_kernel<<<NLINES / 8, 256, 0, stream>>>(x2, w3, b3, out);
}
